// Topological_Loss_45423574122617
// MI455X (gfx1250) — compile-verified, matrix operand order checked by reading, not run on hardware
//
#include <hip/hip_runtime.h>
#include <math.h>

// ---------------------------------------------------------------------------
// Topological (H0 total persistence) loss, fused for MI455X (gfx1250, wave32).
//   K1: Sobel edge magnitude + mean/std(ddof=1) + threshold counts   [stream]
//   K2: threshold resolution (incl. cross-coupled pred->mask fallback)
//   K3: deterministic random top-100 edge-pixel selection (hash radix-select)
//   K4: 112x112 pairwise distances via V_WMMA_F32_16X16X32_F16 + Prim MST
//   K5: final |tp_mask - tp_pred| reduction, * LAM / B
// ---------------------------------------------------------------------------

#define IMG_H 256
#define IMG_W 256
#define NPIX  (IMG_H * IMG_W)
#define BORD  5
#define NPTS  100
#define NPAD  112          // 7 WMMA tiles of 16; pads duplicate a point -> 0-length MST edges
#define NBATCH 128

typedef __attribute__((ext_vector_type(16))) _Float16 v16h;
typedef __attribute__((ext_vector_type(8)))  float    v8f;

// -------------------------------- helpers ----------------------------------

__device__ __forceinline__ float load_px(const float* img, int i, int j, bool sig) {
  if ((unsigned)i >= IMG_H || (unsigned)j >= IMG_W) return 0.0f;   // zero-pad conv input
  float x = img[(i << 8) | j];
  return sig ? 1.0f / (1.0f + expf(-x)) : x;
}

// Sobel magnitude at (i,j); border mask folded in for the pred image.
__device__ __forceinline__ float edge_val(const float* img, int i, int j, bool sig, bool bord) {
  if (bord && (i < BORD || i >= IMG_H - BORD || j < BORD || j >= IMG_W - BORD)) return 0.0f;
  float a00 = load_px(img, i-1, j-1, sig), a01 = load_px(img, i-1, j, sig), a02 = load_px(img, i-1, j+1, sig);
  float a10 = load_px(img, i,   j-1, sig),                                   a12 = load_px(img, i,   j+1, sig);
  float a20 = load_px(img, i+1, j-1, sig), a21 = load_px(img, i+1, j, sig), a22 = load_px(img, i+1, j+1, sig);
  float ex = (a02 - a00) + 2.0f * (a12 - a10) + (a22 - a20);
  float ey = (a20 + 2.0f * a21 + a22) - (a00 + 2.0f * a01 + a02);
  return sqrtf(ex * ex + ey * ey + 1e-12f);
}

__device__ __forceinline__ unsigned hash_key(unsigned seed, unsigned p) {
  unsigned x = p + seed * 0x9E3779B9u + 0x85EBCA6Bu;
  x ^= x >> 16; x *= 0x7FEB352Du;
  x ^= x >> 15; x *= 0x846CA68Bu;
  x ^= x >> 16;
  return x;
}

__device__ __forceinline__ float block_reduce_sum_256(float v, float* buf) {
  int tid = threadIdx.x;
  buf[tid] = v; __syncthreads();
  for (int s = 128; s > 0; s >>= 1) {
    if (tid < s) buf[tid] += buf[tid + s];
    __syncthreads();
  }
  float r = buf[0]; __syncthreads();
  return r;
}

// --------------------------- K1: edge statistics ---------------------------
// grid = 256 (= batch*2), block = 256. blockIdx: b = x>>1, which = x&1 (0=pred,1=mask)
__global__ void edge_stats_kernel(const float* __restrict__ mo,
                                  const float* __restrict__ lb,
                                  float* __restrict__ stats) {
  int bt = blockIdx.x, b = bt >> 1, which = bt & 1;
  const float* img = which ? (lb + (size_t)b * NPIX) : (mo + (size_t)b * NPIX);
  bool sig = (which == 0), bord = (which == 0);
  __shared__ float red[256];
  int tid = threadIdx.x;

  float s = 0.0f, ss = 0.0f;
  for (int p = tid; p < NPIX; p += 256) {
    float e = edge_val(img, p >> 8, p & 255, sig, bord);
    s += e; ss += e * e;
  }
  float S  = block_reduce_sum_256(s,  red);
  float SS = block_reduce_sum_256(ss, red);
  float mean = S * (1.0f / (float)NPIX);
  float var  = fmaxf((SS - S * mean) / (float)(NPIX - 1), 0.0f);   // ddof=1
  float sd   = sqrtf(var);

  float c_hi = 0.0f, c_m = 0.0f;
  for (int p = tid; p < NPIX; p += 256) {
    float e = edge_val(img, p >> 8, p & 255, sig, bord);
    if (e > mean + sd) c_hi += 1.0f;
    if (e > mean)      c_m  += 1.0f;
  }
  float C1 = block_reduce_sum_256(c_hi, red);
  float C2 = block_reduce_sum_256(c_m,  red);
  if (tid == 0) {
    stats[bt * 4 + 0] = mean;
    stats[bt * 4 + 1] = sd;
    stats[bt * 4 + 2] = C1;
    stats[bt * 4 + 3] = C2;
  }
}

// ------------------------- K2: threshold resolution ------------------------
__global__ void resolve_thresholds_kernel(const float* __restrict__ stats,
                                          float* __restrict__ thr) {
  int b = threadIdx.x;
  if (b >= NBATCH) return;
  float pm = stats[(2 * b) * 4 + 0], ps = stats[(2 * b) * 4 + 1];
  float pc_hi = stats[(2 * b) * 4 + 2], pc_m = stats[(2 * b) * 4 + 3];
  bool fb = pc_hi < 5.0f;
  float pthr   = fb ? pm : (pm + ps);
  float pcount = fb ? pc_m : pc_hi;          // post-fallback pred count gates the MASK fallback
  thr[2 * b + 0] = pthr;
  float mm = stats[(2 * b + 1) * 4 + 0], ms = stats[(2 * b + 1) * 4 + 1];
  thr[2 * b + 1] = (pcount < 5.0f) ? mm : (mm + ms);
}

// ------------------- K3: deterministic random point sampling ---------------
// Top-NPTS hashed-key edge pixels via 256-bin radix select + ballot prefix scan.
// grid = 256, block = 256.
__global__ void select_points_kernel(const float* __restrict__ mo,
                                     const float* __restrict__ lb,
                                     const float* __restrict__ thr,
                                     float* __restrict__ pts) {
  int bt = blockIdx.x, b = bt >> 1, which = bt & 1;
  const float* img = which ? (lb + (size_t)b * NPIX) : (mo + (size_t)b * NPIX);
  bool sig = (which == 0), bord = (which == 0);
  float T = thr[bt];

  __shared__ unsigned hist[256];
  __shared__ int whTot[8], wbTot[8], whOff[8], wbOff[8];
  __shared__ int s_cut, s_nab, s_r, s_hiBase, s_bdBase, s_nSel;
  __shared__ float s_p0x, s_p0y;

  int tid = threadIdx.x, lane = tid & 31, w = tid >> 5;
  hist[tid] = 0;
  __syncthreads();

  // Pass A: validity bitmap + key histogram (top 8 bits).
  unsigned vb[8] = {0, 0, 0, 0, 0, 0, 0, 0};
  for (int chunk = 0; chunk < 256; ++chunk) {
    int p = (chunk << 8) | tid;
    float e = edge_val(img, p >> 8, p & 255, sig, bord);
    if (e > T) {
      vb[chunk >> 5] |= 1u << (chunk & 31);
      atomicAdd(&hist[hash_key(bt, p) >> 24], 1u);
    }
  }
  __syncthreads();

  if (tid == 0) {
    int cum = 0, cut = -1, nab = 0;
    for (int bin = 255; bin >= 0; --bin) {
      if (cum + (int)hist[bin] >= NPTS) { cut = bin; nab = cum; break; }
      cum += (int)hist[bin];
    }
    if (cut < 0) nab = cum;                  // fewer than NPTS valid: take all
    s_cut = cut; s_nab = nab;
    s_r = (cut >= 0) ? (NPTS - nab) : 0;
    s_hiBase = 0; s_bdBase = 0;
  }
  __syncthreads();
  int cut = s_cut, nab = s_nab, r = s_r;
  float2* myPts = (float2*)pts + (size_t)bt * NPAD;

  // Pass B: deterministic slot assignment (pixel-index order within the cut bin).
  for (int chunk = 0; chunk < 256; ++chunk) {
    int p = (chunk << 8) | tid;
    bool valid = (vb[chunk >> 5] >> (chunk & 31)) & 1u;
    int bin = valid ? (int)(hash_key(bt, p) >> 24) : -1;
    bool hi = valid && (bin > cut);
    bool bd = valid && (bin == cut);
    unsigned long long mh = __ballot(hi);
    unsigned long long mb = __ballot(bd);
    if (lane == 0) { whTot[w] = __popcll(mh); wbTot[w] = __popcll(mb); }
    __syncthreads();
    if (tid == 0) {
      int a = s_hiBase; for (int k = 0; k < 8; ++k) { whOff[k] = a; a += whTot[k]; } s_hiBase = a;
      a = s_bdBase;     for (int k = 0; k < 8; ++k) { wbOff[k] = a; a += wbTot[k]; } s_bdBase = a;
    }
    __syncthreads();
    unsigned long long below = ((unsigned long long)1 << lane) - 1;
    float fi = (float)(p >> 8), fj = (float)(p & 255);
    if (hi) {
      int slot = whOff[w] + __popcll(mh & below);
      if (slot < NPTS) {
        myPts[slot] = make_float2(fi, fj);
        if (slot == 0) { s_p0x = fi; s_p0y = fj; }
      }
    }
    if (bd) {
      int idx = wbOff[w] + __popcll(mb & below);
      if (idx < r) {
        int slot = nab + idx;
        myPts[slot] = make_float2(fi, fj);
        if (slot == 0) { s_p0x = fi; s_p0y = fj; }
      }
    }
    __syncthreads();
  }

  if (tid == 0) {
    int bdSel = s_bdBase < r ? s_bdBase : r;
    int nsel = s_hiBase + bdSel;
    if (nsel > NPTS) nsel = NPTS;
    if (nsel == 0) { myPts[0] = make_float2(0.0f, 0.0f); s_p0x = 0.0f; s_p0y = 0.0f; nsel = 1; }
    s_nSel = nsel;
  }
  __syncthreads();
  // Pad to NPAD with a duplicate point (adds only zero-length MST edges).
  float2 p0 = make_float2(s_p0x, s_p0y);
  for (int s2 = s_nSel + tid; s2 < NPAD; s2 += 256) myPts[s2] = p0;
}

// ---------------- K4: WMMA distance matrix + Prim total persistence --------
// One wave (32 threads) per (image, cloud). Coordinates are integers in
// [0,255] => exact in FP16; V_WMMA_F32_16X16X32_F16 computes x·y^T exactly
// with F32 accumulation, so D^2 = |xi|^2 + |xj|^2 - 2 xi·xj is bit-exact.
__global__ void mst_wmma_kernel(const float* __restrict__ pts,
                                float* __restrict__ tp) {
  int bt = blockIdx.x;
  const float2* myPts = (const float2*)pts + (size_t)bt * NPAD;
  __shared__ float Px[NPAD], Py[NPAD];
  __shared__ float Dm[NPAD * NPAD];          // 112*112*4 = 50176 B of the 320 KB WGP LDS
  int l = threadIdx.x;

  for (int i = l; i < NPAD; i += 32) { float2 q = myPts[i]; Px[i] = q.x; Py[i] = q.y; }
  __syncthreads();

  // 7x7 tiles of 16x16 via v_wmma_f32_16x16x32_f16 (K=2 live, rest zero).
  for (int ti = 0; ti < 7; ++ti) {
    for (int tj = 0; tj < 7; ++tj) {
      v16h a = {}; v16h bm = {};
      if (l < 16) {                           // lanes 0-15 hold K=0,1 in halves 0,1
        a[0]  = (_Float16)Px[ti * 16 + l]; a[1]  = (_Float16)Py[ti * 16 + l];
        bm[0] = (_Float16)Px[tj * 16 + l]; bm[1] = (_Float16)Py[tj * 16 + l];
      }
      v8f c = {};
      c = __builtin_amdgcn_wmma_f32_16x16x32_f16(false, a, false, bm,
                                                 (short)0, c, false, false);
      int N = tj * 16 + (l & 15);
      float nN = Px[N] * Px[N] + Py[N] * Py[N];
      int mbase = ti * 16 + ((l >> 4) << 3);  // C layout: VGPR v -> M = v (+8 for lanes 16-31)
#pragma unroll
      for (int v = 0; v < 8; ++v) {
        int M = mbase + v;
        float nM = Px[M] * Px[M] + Py[M] * Py[M];
        float d2 = nM + nN - 2.0f * c[v];
        Dm[M * NPAD + N] = sqrtf(fmaxf(d2, 0.0f));
      }
    }
  }
  __syncthreads();

  // Prim's MST: lane l owns entries {l, l+32, l+64, l+96} (e<112).
  const float INF = 1e30f;
  float dist[4]; unsigned tb = 0;
  for (int k = 0; k < 4; ++k) {
    int e = l + 32 * k;
    dist[k] = (e == 0 || e >= NPAD) ? INF : Dm[e];   // dist = D[0], dist[0] = INF
  }
  float tot = 0.0f;
  for (int it = 0; it < NPAD - 1; ++it) {
    float bv = INF; int bi = 0x3fffffff;
    for (int k = 0; k < 4; ++k) {
      if (dist[k] < bv) { bv = dist[k]; bi = l + 32 * k; }
    }
    for (int off = 16; off > 0; off >>= 1) {         // wave32 min+argmin, ties -> lowest index
      float ov = __shfl_xor(bv, off, 32);
      int   oi = __shfl_xor(bi, off, 32);
      if (ov < bv || (ov == bv && oi < bi)) { bv = ov; bi = oi; }
    }
    tot += bv;
    if (bv < 1e29f && (bi & 31) == l) tb |= 1u << (bi >> 5);
    const float* row = &Dm[bi * NPAD];
    for (int k = 0; k < 4; ++k) {
      int e = l + 32 * k;
      if (e >= NPAD || ((tb >> k) & 1u)) dist[k] = INF;
      else dist[k] = fminf(dist[k], row[e]);
    }
  }
  if (l == 0) tp[bt] = tot;
}

// ----------------------------- K5: final loss ------------------------------
__global__ void finalize_kernel(const float* __restrict__ tp, float* __restrict__ out) {
  __shared__ float red[NBATCH];
  int b = threadIdx.x;
  red[b] = fabsf(tp[2 * b + 1] - tp[2 * b + 0]);
  __syncthreads();
  for (int s = NBATCH / 2; s > 0; s >>= 1) {
    if (b < s) red[b] += red[b + s];
    __syncthreads();
  }
  if (b == 0) out[0] = 0.01f * red[0] / (float)NBATCH;
}

// ------------------------------- launcher ----------------------------------
extern "C" void kernel_launch(void* const* d_in, const int* in_sizes, int n_in,
                              void* d_out, int out_size, void* d_ws, size_t ws_size,
                              hipStream_t stream) {
  (void)in_sizes; (void)n_in; (void)out_size; (void)ws_size;
  const float* mo = (const float*)d_in[0];   // [128,1,256,256] f32
  const float* lb = (const float*)d_in[1];   // [128,1,256,256] f32
  float* out = (float*)d_out;                // scalar f32

  // workspace layout (floats): stats[256*4] | thr[256] | tp[256] | pts[256*112*2]
  float* wsf   = (float*)d_ws;
  float* stats = wsf;
  float* thr   = wsf + 1024;
  float* tp    = wsf + 1280;
  float* pts   = wsf + 1536;                 // ~235 KB total

  edge_stats_kernel<<<NBATCH * 2, 256, 0, stream>>>(mo, lb, stats);
  resolve_thresholds_kernel<<<1, NBATCH, 0, stream>>>(stats, thr);
  select_points_kernel<<<NBATCH * 2, 256, 0, stream>>>(mo, lb, thr, pts);
  mst_wmma_kernel<<<NBATCH * 2, 32, 0, stream>>>(pts, tp);
  finalize_kernel<<<1, NBATCH, 0, stream>>>(tp, out);
}
